// EnergyNet_61383672594917
// MI455X (gfx1250) — matrix operand order repaired and checked by reading, not compiled, not run on hardware
//
#include <hip/hip_runtime.h>
#include <cmath>

typedef __attribute__((ext_vector_type(16))) _Float16 v16h;
typedef __attribute__((ext_vector_type(8)))  _Float16 v8h;
typedef __attribute__((ext_vector_type(8)))  float    v8f;

#define C_CH 64
#define NB 8
#define TE 32               // edges per workgroup tile
#define N_LAYERS 3
#define W2S_ELEMS 20480     // 64*320 swizzled f16 weights per layer
#define R_MAX_F 5.0f
#define SQRT3_F 1.7320508075688772f
#define SQRT2_F 1.4142135623730951f
#define INV_SQRT_DEG 0.17677669529663687f   // 1/sqrt(32)
#define PI_F 3.14159265358979323846f

__device__ __forceinline__ float silu_f(float x)    { return x / (1.f + __expf(-x)); }
__device__ __forceinline__ float sigmoid_f(float x) { return 1.f / (1.f + __expf(-x)); }
__device__ __forceinline__ void atomic_add_f32(float* p, float v) {
  __hip_atomic_fetch_add(p, v, __ATOMIC_RELAXED, __HIP_MEMORY_SCOPE_AGENT);
}

// ---------------------------------------------------------------------------
// Edge geometry: y1 = sqrt(3)*unit_vec, emb = bessel * poly cutoff
// ---------------------------------------------------------------------------
__global__ void geom_kernel(const float* __restrict__ pos,
                            const int*   __restrict__ esrc,
                            const int*   __restrict__ edst,
                            const float* __restrict__ shift,
                            const float* __restrict__ cell,
                            const int*   __restrict__ img,
                            float* __restrict__ y1,
                            float* __restrict__ emb, int E) {
  int e = blockIdx.x * blockDim.x + threadIdx.x;
  if (e >= E) return;
  int s = esrc[e], d = edst[e];
  float vx = pos[d*3+0] - pos[s*3+0];
  float vy = pos[d*3+1] - pos[s*3+1];
  float vz = pos[d*3+2] - pos[s*3+2];
  const float* cm = cell + (size_t)img[s] * 9;
  float shx = shift[e*3+0], shy = shift[e*3+1], shz = shift[e*3+2];
  vx += shx*cm[0] + shy*cm[3] + shz*cm[6];
  vy += shx*cm[1] + shy*cm[4] + shz*cm[7];
  vz += shx*cm[2] + shy*cm[5] + shz*cm[8];
  float r  = sqrtf(vx*vx + vy*vy + vz*vz);
  float rs = fmaxf(r, 1e-9f);
  float ri = 1.f / rs;
  y1[e*3+0] = SQRT3_F * vx * ri;
  y1[e*3+1] = SQRT3_F * vy * ri;
  y1[e*3+2] = SQRT3_F * vz * ri;
  // polynomial cutoff p=6: 1 - 28 x^6 + 48 x^7 - 21 x^8
  float x  = r * (1.f / R_MAX_F);
  float x2 = x*x, x4 = x2*x2, x6 = x4*x2;
  float fc = 1.f - 28.f*x6 + 48.f*x6*x - 21.f*x6*x2;
  fc = (x < 1.f) ? fc : 0.f;
  float pref = sqrtf(2.f / R_MAX_F) * ri * fc;
  float warg = PI_F * rs * (1.f / R_MAX_F);
  #pragma unroll
  for (int n = 1; n <= NB; ++n)
    emb[(size_t)e*NB + (n-1)] = pref * __sinf((float)n * warg);
}

// ---------------------------------------------------------------------------
// Pre-swizzle mlp_w2 (f32, [L][64][320]) into per-lane WMMA B-fragment order:
// dest half index = ((nt*2+kt)*32 + lane)*16 + j
// maps to W2[k = kt*32 + (lane>=16)*16 + j][col = nt*16 + (lane&15)]
// Each lane's 16 halves become 32 contiguous bytes -> 2x global_load_b128.
// ---------------------------------------------------------------------------
__global__ void swizzle_w2_kernel(const float* __restrict__ mlp_w2,
                                  _Float16* __restrict__ w2s) {
  int idx = blockIdx.x * blockDim.x + threadIdx.x;
  if (idx >= N_LAYERS * W2S_ELEMS) return;
  int Lr = idx / W2S_ELEMS, r = idx % W2S_ELEMS;
  int nt = r / 1024;  int rem  = r % 1024;
  int kt = rem / 512; int rem2 = rem % 512;
  int l  = rem2 / 16, j = rem2 % 16;
  int row = l & 15, hi = l >> 4;
  int k   = kt*32 + hi*16 + j;
  int col = nt*16 + row;
  w2s[idx] = (_Float16)mlp_w2[(size_t)Lr*64*320 + (size_t)k*320 + col];
}

// ---------------------------------------------------------------------------
// Node feature init: s = w_lin_in[atom_type], v = 0
// vfeat layout: [n][3][64] (component-major for coalescing)
// ---------------------------------------------------------------------------
__global__ void init_kernel(const int* __restrict__ atype,
                            const float* __restrict__ w_lin_in,
                            float* __restrict__ sfeat,
                            float* __restrict__ vfeat, int N) {
  int idx = blockIdx.x * blockDim.x + threadIdx.x;
  if (idx < N * C_CH)
    sfeat[idx] = w_lin_in[atype[idx >> 6] * C_CH + (idx & 63)];
  if (idx < N * C_CH * 3)
    vfeat[idx] = 0.f;
}

__global__ void zero_agg_kernel(float* __restrict__ agg_s,
                                float* __restrict__ agg_v, int N) {
  int idx = blockIdx.x * blockDim.x + threadIdx.x;
  if (idx < N * C_CH)     agg_s[idx] = 0.f;
  if (idx < N * C_CH * 3) agg_v[idx] = 0.f;
}

// ---------------------------------------------------------------------------
// Fused per-layer edge kernel:
//   h = silu(emb @ W1 + b1)     (VALU, K=8)
//   w = h @ W2                   (WMMA f16->f32; A from LDS b128, B from
//                                 pre-swizzled global fragments, L2-resident)
//   CG tensor-product messages + atomic scatter into agg_s / agg_v
// vfeat/agg_v are [n][3][64]: every gather and atomic burst is a contiguous
// 256B range across the 64 threads -> maximal L2 atomic coalescing.
// LDS: 4KB(h f16) + 20KB(out f16) = 24KB -> high occupancy
// ---------------------------------------------------------------------------
__global__ __launch_bounds__(64) void edge_layer_kernel(
    const float* __restrict__ emb, const float* __restrict__ y1g,
    const int*   __restrict__ esrc, const int* __restrict__ edst,
    const float* __restrict__ sfeat, const float* __restrict__ vfeat,
    const float* __restrict__ W1, const float* __restrict__ B1,
    const _Float16* __restrict__ W2s,
    float* __restrict__ agg_s, float* __restrict__ agg_v, int E) {
  __shared__ __align__(16) _Float16 sh_h[TE * 64];      //  4096 B
  __shared__ __align__(16) _Float16 sh_out[TE * 320];   // 20480 B

  const int t    = threadIdx.x;
  const int lane = t & 31;
  const int wave = t >> 5;
  const long e0  = (long)blockIdx.x * TE;

  // hidden layer: one edge per iteration, one channel per thread
  for (int it = 0; it < TE; ++it) {
    long e = e0 + it;
    int  j = t;
    if (e < E) {
      float acc = B1[j];
      #pragma unroll
      for (int k = 0; k < NB; ++k)
        acc += emb[e*NB + k] * W1[k*64 + j];
      sh_h[it*64 + j] = (_Float16)silu_f(acc);
    } else {
      sh_h[it*64 + j] = (_Float16)0.f;
    }
  }
  __syncthreads();

  // WMMA: out[32 x 320] = h[32 x 64] @ W2[64 x 320]
  // wave owns a 16-row M-tile; 20 N-subtiles x 2 K-steps = 40 wmma / wave
  const int m0  = wave * 16;
  const int row = lane & 15;
  const int hi  = (lane >> 4) & 1;   // lane >= 16 ?

  // A fragments (depend only on kt): two contiguous 16B chunks per lane
  v16h afr[2];
  #pragma unroll
  for (int kt = 0; kt < 2; ++kt) {
    const v8h* pa = (const v8h*)&sh_h[(m0 + row)*64 + kt*32 + hi*16];
    v8h alo = pa[0];   // K = kt*32 + hi*16 + 0..7
    v8h ahi = pa[1];   // K = kt*32 + hi*16 + 8..15
    afr[kt] = __builtin_shufflevector(alo, ahi,
        0,1,2,3,4,5,6,7, 8,9,10,11,12,13,14,15);
  }

  const v8h* w2v = (const v8h*)W2s;
  #pragma unroll 4
  for (int nt = 0; nt < 20; ++nt) {
    v8f acc = {};
    #pragma unroll
    for (int kt = 0; kt < 2; ++kt) {
      int fbase = ((nt*2 + kt)*32 + lane)*2;   // v8h index
      v8h blo = w2v[fbase + 0];
      v8h bhi = w2v[fbase + 1];
      v16h b = __builtin_shufflevector(blo, bhi,
          0,1,2,3,4,5,6,7, 8,9,10,11,12,13,14,15);
      acc = __builtin_amdgcn_wmma_f32_16x16x32_f16(
          false, afr[kt], false, b, (short)0, acc, false, false);
    }
    #pragma unroll
    for (int i = 0; i < 8; ++i) {               // D 16x16 f32 layout
      int rr = m0 + hi*8 + i;
      sh_out[rr*320 + nt*16 + row] = (_Float16)acc[i];
    }
  }
  __syncthreads();

  // message phase: one edge per iteration, one channel per thread
  for (int it = 0; it < TE; ++it) {
    long e = e0 + it;
    if (e >= E) break;
    int ch  = t;
    int src = esrc[e], dst = edst[e];
    float yx = y1g[e*3+0], yy = y1g[e*3+1], yz = y1g[e*3+2];
    float ss = sfeat[(size_t)src*C_CH + ch];
    const float* vp = vfeat + (size_t)src*C_CH*3 + ch;   // [n][3][64]
    float vx = vp[0*C_CH], vy = vp[1*C_CH], vz = vp[2*C_CH];
    float w1 = (float)sh_out[it*320 +   0 + ch];
    float w2 = (float)sh_out[it*320 +  64 + ch];
    float w3 = (float)sh_out[it*320 + 128 + ch];
    float w4 = (float)sh_out[it*320 + 192 + ch];
    float w5 = (float)sh_out[it*320 + 256 + ch];
    float vdoty = vx*yx + vy*yy + vz*yz;
    float m_s = w1*ss + w2*vdoty*(1.f/SQRT3_F);
    float cxv = vy*yz - vz*yy;
    float cyv = vz*yx - vx*yz;
    float czv = vx*yy - vy*yx;
    float k3 = w3*ss, k5 = w5*(1.f/SQRT2_F);
    float mvx = k3*yx + w4*vx + k5*cxv;
    float mvy = k3*yy + w4*vy + k5*cyv;
    float mvz = k3*yz + w4*vz + k5*czv;
    atomic_add_f32(&agg_s[(size_t)dst*C_CH + ch], m_s);
    float* ap = agg_v + (size_t)dst*C_CH*3 + ch;         // [n][3][64]
    atomic_add_f32(ap + 0*C_CH, mvx);
    atomic_add_f32(ap + 1*C_CH, mvy);
    atomic_add_f32(ap + 2*C_CH, mvz);
  }
}

// ---------------------------------------------------------------------------
// Node update: gate = sigmoid(agg_s@Wg); s += silu(agg_s@Ws); v += (agg_v@Wv)*gate
// agg_v / vfeat are [n][3][64]
// ---------------------------------------------------------------------------
__global__ void node_update_kernel(const float* __restrict__ agg_s,
                                   const float* __restrict__ agg_v,
                                   const float* __restrict__ Wg,
                                   const float* __restrict__ Ws,
                                   const float* __restrict__ Wv,
                                   float* __restrict__ sfeat,
                                   float* __restrict__ vfeat, int N) {
  int idx = blockIdx.x * blockDim.x + threadIdx.x;
  if (idx >= N * C_CH) return;
  int n = idx >> 6, d = idx & 63;
  const float* as = agg_s + (size_t)n * C_CH;
  const float* av = agg_v + (size_t)n * C_CH * 3;
  float g = 0.f, se = 0.f, ox = 0.f, oy = 0.f, oz = 0.f;
  #pragma unroll 8
  for (int k = 0; k < C_CH; ++k) {
    float a = as[k];
    g  += a * Wg[k*C_CH + d];
    se += a * Ws[k*C_CH + d];
    float wv = Wv[k*C_CH + d];
    ox += av[0*C_CH + k] * wv;
    oy += av[1*C_CH + k] * wv;
    oz += av[2*C_CH + k] * wv;
  }
  float gate = sigmoid_f(g * INV_SQRT_DEG);
  sfeat[idx] += silu_f(se * INV_SQRT_DEG);
  float* vp = vfeat + (size_t)n * C_CH * 3 + d;
  vp[0*C_CH] += ox * INV_SQRT_DEG * gate;
  vp[1*C_CH] += oy * INV_SQRT_DEG * gate;
  vp[2*C_CH] += oz * INV_SQRT_DEG * gate;
}

// ---------------------------------------------------------------------------
// readout: wcomb = w_lin1 @ w_lin2 (64,), then energy[n] = dot(s[n], wcomb)
// ---------------------------------------------------------------------------
__global__ void wcomb_kernel(const float* __restrict__ w_lin1,
                             const float* __restrict__ w_lin2,
                             float* __restrict__ wcomb) {
  int c = threadIdx.x;
  if (c >= C_CH) return;
  float acc = 0.f;
  #pragma unroll
  for (int j = 0; j < 16; ++j) acc += w_lin1[c*16 + j] * w_lin2[j];
  wcomb[c] = acc;
}

__global__ void energy_kernel(const float* __restrict__ sfeat,
                              const float* __restrict__ wcomb,
                              float* __restrict__ out, int N) {
  int n = blockIdx.x * blockDim.x + threadIdx.x;
  if (n >= N) return;
  const float* sp = sfeat + (size_t)n * C_CH;
  float acc = 0.f;
  #pragma unroll 8
  for (int c = 0; c < C_CH; ++c) acc += sp[c] * wcomb[c];
  out[n] = acc;
}

// ---------------------------------------------------------------------------
extern "C" void kernel_launch(void* const* d_in, const int* in_sizes, int n_in,
                              void* d_out, int out_size, void* d_ws, size_t ws_size,
                              hipStream_t stream) {
  const int*   atom_type = (const int*)  d_in[0];
  const float* atom_pos  = (const float*)d_in[1];
  const int*   edge_src  = (const int*)  d_in[2];
  const int*   edge_dst  = (const int*)  d_in[3];
  const float* cellshift = (const float*)d_in[4];
  const float* cell      = (const float*)d_in[5];
  const int*   image_idx = (const int*)  d_in[6];
  const float* w_lin_in  = (const float*)d_in[7];
  const float* mlp_w1    = (const float*)d_in[8];
  const float* mlp_b1    = (const float*)d_in[9];
  const float* mlp_w2    = (const float*)d_in[10];
  const float* w_self_s  = (const float*)d_in[11];
  const float* w_self_v  = (const float*)d_in[12];
  const float* w_gate    = (const float*)d_in[13];
  const float* w_lin1    = (const float*)d_in[14];
  const float* w_lin2    = (const float*)d_in[15];
  float* out = (float*)d_out;

  const int N = in_sizes[0];
  const int E = in_sizes[2];

  // workspace layout (f32 then f16 region; all offsets 16B aligned)
  float* wsf   = (float*)d_ws;
  float* y1    = wsf;                               // E*3
  float* emb   = y1    + (size_t)E * 3;             // E*8
  float* sfeat = emb   + (size_t)E * 8;             // N*64
  float* vfeat = sfeat + (size_t)N * C_CH;          // N*3*64
  float* agg_s = vfeat + (size_t)N * C_CH * 3;      // N*64
  float* agg_v = agg_s + (size_t)N * C_CH;          // N*3*64
  float* wcomb = agg_v + (size_t)N * C_CH * 3;      // 64
  _Float16* w2s_all = (_Float16*)(wcomb + 64);      // 3*20480 halves

  const int bs = 256;
  swizzle_w2_kernel<<<(N_LAYERS * W2S_ELEMS + bs - 1) / bs, bs, 0, stream>>>(
      mlp_w2, w2s_all);
  geom_kernel<<<(E + bs - 1) / bs, bs, 0, stream>>>(
      atom_pos, edge_src, edge_dst, cellshift, cell, image_idx, y1, emb, E);
  init_kernel<<<(N * C_CH * 3 + bs - 1) / bs, bs, 0, stream>>>(
      atom_type, w_lin_in, sfeat, vfeat, N);

  for (int L = 0; L < N_LAYERS; ++L) {
    zero_agg_kernel<<<(N * C_CH * 3 + bs - 1) / bs, bs, 0, stream>>>(agg_s, agg_v, N);
    edge_layer_kernel<<<(E + TE - 1) / TE, 64, 0, stream>>>(
        emb, y1, edge_src, edge_dst, sfeat, vfeat,
        mlp_w1 + (size_t)L * NB * 64,
        mlp_b1 + (size_t)L * 64,
        w2s_all + (size_t)L * W2S_ELEMS,
        agg_s, agg_v, E);
    node_update_kernel<<<(N * C_CH + bs - 1) / bs, bs, 0, stream>>>(
        agg_s, agg_v,
        w_gate   + (size_t)L * C_CH * C_CH,
        w_self_s + (size_t)L * C_CH * C_CH,
        w_self_v + (size_t)L * C_CH * C_CH,
        sfeat, vfeat, N);
  }

  wcomb_kernel<<<1, 64, 0, stream>>>(w_lin1, w_lin2, wcomb);
  energy_kernel<<<(N + bs - 1) / bs, bs, 0, stream>>>(sfeat, wcomb, out, N);
}
